// SelectiveSSM_51127290692381
// MI455X (gfx1250) — matrix operand order repaired and checked by reading, not compiled, not run on hardware
//
#include <hip/hip_runtime.h>
#include <hip/hip_bf16.h>
#include <cstdint>

#define D_MODEL 1024
#define D_STATE 16
#define DT_RANK 64
#define BATCH   4
#define SEQ     2048
#define NC      8            // scan chunks along L
#define LC      (SEQ / NC)   // 256
#define M_TOTAL (BATCH * SEQ)
#define LOG2E   1.4426950408889634f

typedef __attribute__((ext_vector_type(16))) _Float16 v16h;
typedef __attribute__((ext_vector_type(8)))  _Float16 v8h;
typedef __attribute__((ext_vector_type(8)))  float    v8f;
typedef __attribute__((ext_vector_type(4)))  float    v4f;

// ---------------------------------------------------------------- f32 -> f16
__global__ void k_f32_to_f16(const float* __restrict__ src,
                             _Float16* __restrict__ dst, int n) {
  int i = blockIdx.x * blockDim.x + threadIdx.x;
  if (i < n) dst[i] = (_Float16)src[i];
}

// A-operand (16x32 f16): lane l holds M=l%16; K = (l/16)*16 + {0..15} (contiguous, same as B).
// B-operand (32x16 f16): lane l holds N=l%16; K = (l/16)*16 + {0..15}.
// C/D (16x16 f32): lane l holds N=l%16; VGPR v holds M = v + 8*(l/16).

// ------------------------------------------------- fused x-projections (WMMA)
// r = x @ x_proj_w^T (f16 out), Bm = x @ B_proj_w^T, Cm = x @ C_proj_w^T (f32)
// One wave per 16-row M-tile; 6 N-tiles (4 for r, 1 Bm, 1 Cm); K=1024 in 32s.
__global__ __launch_bounds__(128) void k_proj(
    const float* __restrict__ x,
    const _Float16* __restrict__ wx,   // 64 x 1024
    const _Float16* __restrict__ wB,   // 16 x 1024
    const _Float16* __restrict__ wC,   // 16 x 1024
    _Float16* __restrict__ r,          // 8192 x 64
    float* __restrict__ Bm,            // 8192 x 16
    float* __restrict__ Cm) {          // 8192 x 16
  const int lane  = threadIdx.x & 31;
  const int mtile = (blockIdx.x * blockDim.x + threadIdx.x) >> 5;  // 0..511
  const int lr = lane & 15;
  const int lh = lane >> 4;
  const int kbA = lh * 16;
  const int kbB = lh * 16;

  const float* xrow = x + (size_t)(mtile * 16 + lr) * D_MODEL;
  const _Float16* wp[6];
#pragma unroll
  for (int t = 0; t < 4; ++t)
    wp[t] = wx + (size_t)(t * 16 + lr) * D_MODEL + kbB;
  wp[4] = wB + (size_t)lr * D_MODEL + kbB;
  wp[5] = wC + (size_t)lr * D_MODEL + kbB;

  v8f acc[6];
#pragma unroll
  for (int t = 0; t < 6; ++t) acc[t] = {};

  for (int k = 0; k < D_MODEL; k += 32) {
    v8f xlo = *(const v8f*)(xrow + k + kbA);
    v8f xhi = *(const v8f*)(xrow + k + kbA + 8);
    v8h alo = __builtin_convertvector(xlo, v8h);
    v8h ahi = __builtin_convertvector(xhi, v8h);
    v16h a = __builtin_shufflevector(alo, ahi, 0, 1, 2, 3, 4, 5, 6, 7,
                                     8, 9, 10, 11, 12, 13, 14, 15);
#pragma unroll
    for (int t = 0; t < 6; ++t) {
      v16h b = *(const v16h*)(wp[t] + k);
      acc[t] = __builtin_amdgcn_wmma_f32_16x16x32_f16(
          false, a, false, b, (short)0, acc[t], false, false);
    }
  }

#pragma unroll
  for (int v = 0; v < 8; ++v) {
    const int m = mtile * 16 + v + 8 * lh;
#pragma unroll
    for (int t = 0; t < 4; ++t)
      r[(size_t)m * DT_RANK + t * 16 + lr] = (_Float16)acc[t][v];
    Bm[(size_t)m * D_STATE + lr] = acc[4][v];
    Cm[(size_t)m * D_STATE + lr] = acc[5][v];
  }
}

// --------------------------------- delta = clamp(softplus(r @ dt_w^T + b))
// M=8192, N=1024, K=64. One wave per 16x16 tile -> 32768 waves.
__global__ __launch_bounds__(256) void k_delta(
    const _Float16* __restrict__ r,     // 8192 x 64
    const _Float16* __restrict__ wdt,   // 1024 x 64
    const float* __restrict__ bias,     // 1024
    float* __restrict__ delta) {        // 8192 x 1024
  const int lane = threadIdx.x & 31;
  const int wave = (blockIdx.x * blockDim.x + threadIdx.x) >> 5;  // 0..32767
  const int mtile = wave >> 6;
  const int ntile = wave & 63;
  const int lr = lane & 15;
  const int lh = lane >> 4;
  const int kbA = lh * 16;
  const int kbB = lh * 16;

  const _Float16* ap = r + (size_t)(mtile * 16 + lr) * DT_RANK;
  const _Float16* bp = wdt + (size_t)(ntile * 16 + lr) * DT_RANK + kbB;

  v8f acc = {};
#pragma unroll
  for (int s = 0; s < 2; ++s) {
    const int k = s * 32;
    v8h alo = *(const v8h*)(ap + k + kbA);
    v8h ahi = *(const v8h*)(ap + k + kbA + 8);
    v16h a = __builtin_shufflevector(alo, ahi, 0, 1, 2, 3, 4, 5, 6, 7,
                                     8, 9, 10, 11, 12, 13, 14, 15);
    v16h b = *(const v16h*)(bp + k);
    acc = __builtin_amdgcn_wmma_f32_16x16x32_f16(
        false, a, false, b, (short)0, acc, false, false);
  }

  const float bz = bias[ntile * 16 + lr];
#pragma unroll
  for (int v = 0; v < 8; ++v) {
    float z = acc[v] + bz;
    float sp = (z > 20.f) ? z : __logf(1.f + __expf(z));
    sp = fminf(fmaxf(sp, 0.001f), 0.1f);
    delta[(size_t)(mtile * 16 + v + 8 * lh) * D_MODEL + ntile * 16 + lr] = sp;
  }
}

// ------------------------------------- phase 1: per-chunk local scan (h0 = 0)
// 4 consecutive blocks share one (b, chunk); block covers 256 consecutive d.
// Bm rows for the whole chunk staged to LDS once (16 KB), then served as
// conflict-free broadcast ds_load_b128 in the per-step loop.
__global__ __launch_bounds__(256) void k_scan1(
    const float* __restrict__ delta, const float* __restrict__ x,
    const float* __restrict__ Bm, const float* __restrict__ A_log,
    float* __restrict__ hend, float* __restrict__ dsum_out) {
  __shared__ float sB[LC * D_STATE];  // 16 KB

  const int tid = threadIdx.x;
  const int bc = blockIdx.x >> 2;             // 0..31  (b*NC + c)
  const int c = bc & (NC - 1);
  const int b = bc >> 3;
  const int d = ((blockIdx.x & 3) << 8) + tid;
  const int gid = bc * D_MODEL + d;

  const size_t row0 = (size_t)b * SEQ + (size_t)c * LC;

  // cooperative stage of Bm chunk rows into LDS (coalesced b128 loads)
  {
    const float* bsrc = Bm + row0 * D_STATE;
#pragma unroll
    for (int i = 0; i < (LC * D_STATE) / (256 * 4); ++i) {
      const int e = (i * 256 + tid) * 4;
      *(v4f*)(sB + e) = *(const v4f*)(bsrc + e);
    }
  }
  __syncthreads();

  float A2[D_STATE];
#pragma unroll
  for (int n = 0; n < D_STATE; ++n)
    A2[n] = -__expf(A_log[d * D_STATE + n]) * LOG2E;

  float h[D_STATE];
#pragma unroll
  for (int n = 0; n < D_STATE; ++n) h[n] = 0.f;
  float ds = 0.f;

  const float* dp = delta + row0 * D_MODEL + d;
  const float* xp = x + row0 * D_MODEL + d;

  for (int t = 0; t < LC; ++t) {
    const float dlt = dp[(size_t)t * D_MODEL];
    const float xv = xp[(size_t)t * D_MODEL];
    const float dx = dlt * xv;
    ds += dlt;
    float Bv[D_STATE];
    *(v4f*)(Bv + 0)  = *(const v4f*)(sB + t * D_STATE + 0);
    *(v4f*)(Bv + 4)  = *(const v4f*)(sB + t * D_STATE + 4);
    *(v4f*)(Bv + 8)  = *(const v4f*)(sB + t * D_STATE + 8);
    *(v4f*)(Bv + 12) = *(const v4f*)(sB + t * D_STATE + 12);
#pragma unroll
    for (int n = 0; n < D_STATE; ++n)
      h[n] = __builtin_amdgcn_exp2f(dlt * A2[n]) * h[n] + dx * Bv[n];
  }

#pragma unroll
  for (int n = 0; n < D_STATE; ++n) hend[(size_t)gid * D_STATE + n] = h[n];
  dsum_out[gid] = ds;
}

// ------------------- phase 2: sequential chunk combine (tiny) -> h_init/chunk
__global__ __launch_bounds__(256) void k_combine(
    const float* __restrict__ hend, const float* __restrict__ dsum,
    const float* __restrict__ A_log, float* __restrict__ hinit) {
  const int gid = blockIdx.x * blockDim.x + threadIdx.x;  // 0..4095
  const int d = gid & (D_MODEL - 1);
  const int b = gid >> 10;

  float A2[D_STATE];
#pragma unroll
  for (int n = 0; n < D_STATE; ++n)
    A2[n] = -__expf(A_log[d * D_STATE + n]) * LOG2E;

  float h[D_STATE];
#pragma unroll
  for (int n = 0; n < D_STATE; ++n) h[n] = 0.f;

  for (int c = 0; c < NC; ++c) {
    const size_t idx = (size_t)(b * NC + c) * D_MODEL + d;
#pragma unroll
    for (int n = 0; n < D_STATE; ++n) hinit[idx * D_STATE + n] = h[n];
    if (c < NC - 1) {
      const float s = dsum[idx];
#pragma unroll
      for (int n = 0; n < D_STATE; ++n)
        h[n] = __builtin_amdgcn_exp2f(s * A2[n]) * h[n] +
               hend[idx * D_STATE + n];
    }
  }
}

// ---------------------- phase 3: per-chunk scan with correct h_init; emit y
// Bm and Cm chunk rows staged to LDS (32 KB total).
__global__ __launch_bounds__(256) void k_scan2(
    const float* __restrict__ delta, const float* __restrict__ x,
    const float* __restrict__ Bm, const float* __restrict__ Cm,
    const float* __restrict__ A_log, const float* __restrict__ Dvec,
    const float* __restrict__ hinit, float* __restrict__ out) {
  __shared__ float sB[LC * D_STATE];  // 16 KB
  __shared__ float sC[LC * D_STATE];  // 16 KB

  const int tid = threadIdx.x;
  const int bc = blockIdx.x >> 2;             // 0..31  (b*NC + c)
  const int c = bc & (NC - 1);
  const int b = bc >> 3;
  const int d = ((blockIdx.x & 3) << 8) + tid;
  const int gid = bc * D_MODEL + d;

  const size_t row0 = (size_t)b * SEQ + (size_t)c * LC;

  {
    const float* bsrc = Bm + row0 * D_STATE;
    const float* csrc = Cm + row0 * D_STATE;
#pragma unroll
    for (int i = 0; i < (LC * D_STATE) / (256 * 4); ++i) {
      const int e = (i * 256 + tid) * 4;
      *(v4f*)(sB + e) = *(const v4f*)(bsrc + e);
      *(v4f*)(sC + e) = *(const v4f*)(csrc + e);
    }
  }
  __syncthreads();

  float A2[D_STATE];
#pragma unroll
  for (int n = 0; n < D_STATE; ++n)
    A2[n] = -__expf(A_log[d * D_STATE + n]) * LOG2E;

  float h[D_STATE];
#pragma unroll
  for (int n = 0; n < D_STATE; ++n) h[n] = hinit[(size_t)gid * D_STATE + n];
  const float Dd = Dvec[d];

  const float* dp = delta + row0 * D_MODEL + d;
  const float* xp = x + row0 * D_MODEL + d;
  float* op = out + row0 * D_MODEL + d;

  for (int t = 0; t < LC; ++t) {
    const float dlt = dp[(size_t)t * D_MODEL];
    const float xv = xp[(size_t)t * D_MODEL];
    const float dx = dlt * xv;
    float Bv[D_STATE], Cv[D_STATE];
    *(v4f*)(Bv + 0)  = *(const v4f*)(sB + t * D_STATE + 0);
    *(v4f*)(Bv + 4)  = *(const v4f*)(sB + t * D_STATE + 4);
    *(v4f*)(Bv + 8)  = *(const v4f*)(sB + t * D_STATE + 8);
    *(v4f*)(Bv + 12) = *(const v4f*)(sB + t * D_STATE + 12);
    *(v4f*)(Cv + 0)  = *(const v4f*)(sC + t * D_STATE + 0);
    *(v4f*)(Cv + 4)  = *(const v4f*)(sC + t * D_STATE + 4);
    *(v4f*)(Cv + 8)  = *(const v4f*)(sC + t * D_STATE + 8);
    *(v4f*)(Cv + 12) = *(const v4f*)(sC + t * D_STATE + 12);
    float y = 0.f;
#pragma unroll
    for (int n = 0; n < D_STATE; ++n) {
      h[n] = __builtin_amdgcn_exp2f(dlt * A2[n]) * h[n] + dx * Bv[n];
      y = fmaf(h[n], Cv[n], y);
    }
    op[(size_t)t * D_MODEL] = fmaf(Dd, xv, y);
  }
}

// ---------------------------------------------------------------------------
extern "C" void kernel_launch(void* const* d_in, const int* in_sizes, int n_in,
                              void* d_out, int out_size, void* d_ws,
                              size_t ws_size, hipStream_t stream) {
  const float* x     = (const float*)d_in[0];  // (4,2048,1024)
  const float* A_log = (const float*)d_in[1];  // (1024,16)
  const float* Dv    = (const float*)d_in[2];  // (1024,)
  const float* dt_w  = (const float*)d_in[3];  // (1024,64)
  const float* dt_b  = (const float*)d_in[4];  // (1024,)
  const float* xw    = (const float*)d_in[5];  // (64,1024)
  const float* Bw    = (const float*)d_in[6];  // (16,1024)
  const float* Cw    = (const float*)d_in[7];  // (16,1024)
  float* out = (float*)d_out;

  char* ws = (char*)d_ws;
  _Float16* wx_h  = (_Float16*)(ws + 0);         // 128 KB
  _Float16* wdt_h = (_Float16*)(ws + 131072);    // 128 KB
  _Float16* wB_h  = (_Float16*)(ws + 262144);    // 32 KB
  _Float16* wC_h  = (_Float16*)(ws + 294912);    // 32 KB
  _Float16* r_h   = (_Float16*)(ws + 327680);    // 1 MB  (8192x64 f16)
  float* delta    = (float*)(ws + 1376256);      // 32 MB (8192x1024 f32)
  float* Bm       = (float*)(ws + 34930688);     // 0.5 MB
  float* Cm       = (float*)(ws + 35454976);     // 0.5 MB
  float* hend     = (float*)(ws + 35979264);     // 2 MB (4*8*1024*16)
  float* dsum     = (float*)(ws + 38076416);     // 128 KB
  float* hinit    = (float*)(ws + 38207488);     // 2 MB
  // total ws use ~40.3 MB

  k_f32_to_f16<<<(65536 + 255) / 256, 256, 0, stream>>>(xw, wx_h, 65536);
  k_f32_to_f16<<<(65536 + 255) / 256, 256, 0, stream>>>(dt_w, wdt_h, 65536);
  k_f32_to_f16<<<(16384 + 255) / 256, 256, 0, stream>>>(Bw, wB_h, 16384);
  k_f32_to_f16<<<(16384 + 255) / 256, 256, 0, stream>>>(Cw, wC_h, 16384);

  k_proj<<<128, 128, 0, stream>>>(x, wx_h, wB_h, wC_h, r_h, Bm, Cm);
  k_delta<<<4096, 256, 0, stream>>>(r_h, wdt_h, dt_b, delta);

  k_scan1<<<128, 256, 0, stream>>>(delta, x, Bm, A_log, hend, dsum);
  k_combine<<<16, 256, 0, stream>>>(hend, dsum, A_log, hinit);
  k_scan2<<<128, 256, 0, stream>>>(delta, x, Bm, Cm, A_log, Dv, hinit, out);
}
